// _ScaledDotProductAttention_5892695130577
// MI455X (gfx1250) — matrix operand order repaired and checked by reading, not compiled, not run on hardware
//
#include <hip/hip_runtime.h>
#include <hip/hip_bf16.h>

// ---- CDNA5 WMMA vector types ----
typedef __attribute__((ext_vector_type(16))) __bf16 v16bf;
typedef __attribute__((ext_vector_type(8)))  float  v8f;

#define B_      16
#define L_      2048
#define D_      128
#define BM      128            // query rows per block (16 per wave x 8 waves)
#define BN      32             // key rows per tile
#define NTILES  (L_ / BN)      // 64

// LDS strides, padded to avoid bank conflicts (64 banks x 4B)
#define KSTRIDE32  69          // K tile: u32 (bf16-pair) stride per key row (>=64)
#define VTSTRIDE16 42          // V^T tile: bf16 stride per d row (>=32)
#define PSTRIDE16  36          // P stage: bf16 stride per row (>=32)

#define KBUF (BN * KSTRIDE32)
#define VBUF (D_ * VTSTRIDE16)

// f32x2 -> packed bf16x2. Prefer the CDNA5 packed convert (1 VALU op);
// fall back to manual RNE if the builtin isn't declared for this target.
__device__ __forceinline__ unsigned pack2bf(float a, float b) {
#if defined(__has_builtin) && __has_builtin(__builtin_amdgcn_cvt_pk_bf16_f32)
  typedef __attribute__((ext_vector_type(2))) __bf16 v2bf;
  union { v2bf v; unsigned u; } r;
  r.v = __builtin_amdgcn_cvt_pk_bf16_f32(a, b);
  return r.u;
#else
  unsigned ua = __float_as_uint(a); ua += 0x7FFFu + ((ua >> 16) & 1u);
  unsigned ub = __float_as_uint(b); ub += 0x7FFFu + ((ub >> 16) & 1u);
  return (ua >> 16) | (ub & 0xFFFF0000u);
#endif
}

union BF16Frag { v16bf v; unsigned u[8]; };

__global__ __launch_bounds__(256)
void flash_attn_bf16_wmma(const float* __restrict__ q,
                          const float* __restrict__ k,
                          const float* __restrict__ v,
                          float* __restrict__ out) {
  __shared__ unsigned       lds_k [2 * KBUF];              // [buf][key][d/2] bf16 pairs
  __shared__ unsigned short lds_vt[2 * VBUF];              // [buf][d][key]  (V transposed)
  __shared__ unsigned short lds_p [8 * 16 * PSTRIDE16];    // per-wave P restage

  const int tid  = threadIdx.x;
  const int wave = tid >> 5;
  const int lane = tid & 31;
  const int h    = lane >> 4;      // 16-lane half
  const int m    = lane & 15;

  const int b  = blockIdx.x >> 4;      // 16 query tiles per batch
  const int qt = blockIdx.x & 15;
  const int q0 = qt * BM + wave * 16;  // this wave's first query row

  // scores kept in log2 domain: fold log2(e) into the 1/sqrt(d) scale so
  // every softmax exponential is a bare v_exp_f32 (TRANS pipe).
  const float qscale = 0.08838834764831845f * 1.4426950408889634f;

  const size_t bofs = (size_t)b * L_ * D_;
  const float* qb = q + bofs;
  const float* kb = k + bofs;
  const float* vb = v + bofs;
  float*       ob = out + bofs;

  // per-thread cooperative-load coordinates (64 bf16-pairs per row)
  const int ld_key = tid >> 3;               // 0..31  (key row, 8 threads per row)
  // each thread covers pairs  dp = (tid&7) + ii*8  of its row (ii = 0..7)

  // ---- Q A-fragments (16x32 bf16 per K-chunk), pre-scaled ----
  // 16-bit A layout: lane half h, row m; pair j -> K=16h+2j (contiguous, matches B)
  BF16Frag qf[4];
#pragma unroll
  for (int kc = 0; kc < 4; ++kc) {
#pragma unroll
    for (int j = 0; j < 8; ++j) {
      int d = kc * 32 + 16 * h + 2 * j;
      const float* p = qb + (size_t)(q0 + m) * D_ + d;
      qf[kc].u[j] = pack2bf(p[0] * qscale, p[1] * qscale);
    }
  }

  // ---- accumulators: O (16x128 f32 = 8 groups of v8f), running max/sum ----
  v8f acc[8];
  const v8f vzero = {0.f, 0.f, 0.f, 0.f, 0.f, 0.f, 0.f, 0.f};
#pragma unroll
  for (int g = 0; g < 8; ++g) acc[g] = vzero;
  float mrow[8], lrow[8];
#pragma unroll
  for (int r = 0; r < 8; ++r) { mrow[r] = -1e30f; lrow[r] = 0.0f; }

  // ---- prologue: load + convert + store tile 0 into buffer 0 ----
#pragma unroll
  for (int ii = 0; ii < 8; ++ii) {
    int dp = (tid & 7) + ii * 8;
    const float2 kv = *(const float2*)(kb + (size_t)ld_key * D_ + dp * 2);
    const float2 vv = *(const float2*)(vb + (size_t)ld_key * D_ + dp * 2);
    lds_k[ld_key * KSTRIDE32 + dp] = pack2bf(kv.x, kv.y);
    unsigned pv = pack2bf(vv.x, vv.y);
    lds_vt[(dp * 2 + 0) * VTSTRIDE16 + ld_key] = (unsigned short)pv;
    lds_vt[(dp * 2 + 1) * VTSTRIDE16 + ld_key] = (unsigned short)(pv >> 16);
  }
  __syncthreads();

  for (int t = 0; t < NTILES; ++t) {
    const int cur = t & 1;
    const unsigned*       kbuf  = lds_k  + cur * KBUF;
    const unsigned short* vtbuf = lds_vt + cur * VBUF;

    // ---- issue raw global loads for tile t+1 NOW (latency hides under WMMA) ----
    float2 kreg[8], vreg[8];
    const bool more = (t + 1) < NTILES;
    if (more) {
      const float* kn = kb + (size_t)(t + 1) * BN * D_ + (size_t)ld_key * D_;
      const float* vn = vb + (size_t)(t + 1) * BN * D_ + (size_t)ld_key * D_;
#pragma unroll
      for (int ii = 0; ii < 8; ++ii) {
        int dp = (tid & 7) + ii * 8;
        kreg[ii] = *(const float2*)(kn + dp * 2);
        vreg[ii] = *(const float2*)(vn + dp * 2);
      }
      if (t + 2 < NTILES) {  // pull tile t+2 toward L2
        __builtin_prefetch((const char*)(kb + (size_t)(t + 2) * BN * D_) + tid * 64, 0, 1);
        __builtin_prefetch((const char*)(vb + (size_t)(t + 2) * BN * D_) + tid * 64, 0, 1);
      }
    }

    // ---- S = Q * K^T : two 16-key groups, four K=32 chunks each ----
    v8f s[2];
#pragma unroll
    for (int nt = 0; nt < 2; ++nt) {
      v8f c = vzero;
#pragma unroll
      for (int kc = 0; kc < 4; ++kc) {
        BF16Frag bf;   // B layout: lane = column n, h picks K half, pairs contiguous
#pragma unroll
        for (int j = 0; j < 8; ++j)
          bf.u[j] = kbuf[(nt * 16 + m) * KSTRIDE32 + kc * 16 + h * 8 + j];
        c = __builtin_amdgcn_wmma_f32_16x16x32_bf16(false, qf[kc].v, false, bf.v,
                                                    (short)0, c, false, false);
      }
      s[nt] = c;
    }

    // ---- online softmax (log2 domain); rows live across 16-lane halves ----
    float mx[8];
#pragma unroll
    for (int r = 0; r < 8; ++r) mx[r] = fmaxf(s[0][r], s[1][r]);
#pragma unroll
    for (int mk = 1; mk <= 8; mk <<= 1)
#pragma unroll
      for (int r = 0; r < 8; ++r) mx[r] = fmaxf(mx[r], __shfl_xor(mx[r], mk, 32));

    // rescale only when the running max actually changed (wave-uniform branch)
    bool ch = false;
#pragma unroll
    for (int r = 0; r < 8; ++r) ch = ch || (mx[r] > mrow[r]);
    if (__any((int)ch)) {
      float alpha[8];
#pragma unroll
      for (int r = 0; r < 8; ++r) {
        float nmv = fmaxf(mx[r], mrow[r]);
        alpha[r]  = __builtin_exp2f(mrow[r] - nmv);   // v_exp_f32
        mrow[r]   = nmv;
        lrow[r]  *= alpha[r];
      }
#pragma unroll
      for (int g = 0; g < 8; ++g)
#pragma unroll
        for (int r = 0; r < 8; ++r) acc[g][r] *= alpha[r];
    }

#pragma unroll
    for (int nt = 0; nt < 2; ++nt)
#pragma unroll
      for (int r = 0; r < 8; ++r)
        s[nt][r] = __builtin_exp2f(s[nt][r] - mrow[r]);  // bare v_exp_f32
    float rs[8];
#pragma unroll
    for (int r = 0; r < 8; ++r) rs[r] = s[0][r] + s[1][r];
#pragma unroll
    for (int mk = 1; mk <= 8; mk <<= 1)
#pragma unroll
      for (int r = 0; r < 8; ++r) rs[r] += __shfl_xor(rs[r], mk, 32);
#pragma unroll
    for (int r = 0; r < 8; ++r) lrow[r] += rs[r];

    // ---- restage P (C layout -> A layout) through wave-private LDS ----
    unsigned short* pw = lds_p + wave * 16 * PSTRIDE16;
#pragma unroll
    for (int r = 0; r < 8; ++r) {
      unsigned pv = pack2bf(s[0][r], s[1][r]);           // one cvt, two halves
      pw[(r + 8 * h) * PSTRIDE16 + 0 * 16 + m] = (unsigned short)pv;
      pw[(r + 8 * h) * PSTRIDE16 + 1 * 16 + m] = (unsigned short)(pv >> 16);
    }
    asm volatile("s_wait_dscnt 0" ::: "memory");   // DS in-order; fence compiler too
    BF16Frag pf;
    {
      const unsigned* pw32 = (const unsigned*)pw;
#pragma unroll
      for (int j = 0; j < 8; ++j) {
        int off = 8 * h + j;   // K-pair index per A layout
        pf.u[j] = pw32[m * (PSTRIDE16 / 2) + off];
      }
    }

    // ---- O += P * V (8 output column groups of 16) ----
    const unsigned* vt32 = (const unsigned*)vtbuf;
#pragma unroll
    for (int g = 0; g < 8; ++g) {
      BF16Frag vf;   // B layout from V^T: lane = out column d, K(=key) pairs contiguous
#pragma unroll
      for (int j = 0; j < 8; ++j)
        vf.u[j] = vt32[(g * 16 + m) * (VTSTRIDE16 / 2) + h * 8 + j];
      acc[g] = __builtin_amdgcn_wmma_f32_16x16x32_bf16(false, pf.v, false, vf.v,
                                                       (short)0, acc[g], false, false);
    }

    // ---- convert + store tile t+1 into the other buffer, then one barrier ----
    if (more) {
      unsigned*       kwr  = lds_k  + (cur ^ 1) * KBUF;
      unsigned short* vtwr = lds_vt + (cur ^ 1) * VBUF;
#pragma unroll
      for (int ii = 0; ii < 8; ++ii) {
        int dp = (tid & 7) + ii * 8;
        kwr[ld_key * KSTRIDE32 + dp] = pack2bf(kreg[ii].x, kreg[ii].y);
        unsigned pv = pack2bf(vreg[ii].x, vreg[ii].y);
        vtwr[(dp * 2 + 0) * VTSTRIDE16 + ld_key] = (unsigned short)pv;
        vtwr[(dp * 2 + 1) * VTSTRIDE16 + ld_key] = (unsigned short)(pv >> 16);
      }
    }
    __syncthreads();
  }

  // ---- normalize by row sums and store ----
  float inv[8];
#pragma unroll
  for (int r = 0; r < 8; ++r) inv[r] = 1.0f / lrow[r];
#pragma unroll
  for (int g = 0; g < 8; ++g)
#pragma unroll
    for (int r = 0; r < 8; ++r) {
      int row = q0 + r + 8 * h;
      int col = g * 16 + m;
      ob[(size_t)row * D_ + col] = acc[g][r] * inv[r];
    }
}

extern "C" void kernel_launch(void* const* d_in, const int* in_sizes, int n_in,
                              void* d_out, int out_size, void* d_ws, size_t ws_size,
                              hipStream_t stream) {
  const float* q = (const float*)d_in[0];
  const float* k = (const float*)d_in[1];
  const float* v = (const float*)d_in[2];
  float* out = (float*)d_out;
  (void)in_sizes; (void)n_in; (void)out_size; (void)d_ws; (void)ws_size;
  dim3 grid(B_ * (L_ / BM));   // 16 batches x 16 query tiles = 256 blocks
  dim3 block(256);             // 8 waves (wave32)
  hipLaunchKernelGGL(flash_attn_bf16_wmma, grid, block, 0, stream, q, k, v, out);
}
